// bMamba_26164940767892
// MI455X (gfx1250) — matrix operand order repaired and checked by reading, not compiled, not run on hardware
//
#include <hip/hip_runtime.h>

// ---------------- problem constants ----------------
#define BATCH  4
#define SEQ    1024
#define BL     (BATCH*SEQ)   // 4096 rows everywhere
#define DM     768           // d_model
#define DI     1536          // d_inner
#define DST    16            // d_state
#define DCONV  4
#define DTR    48            // dt_rank
#define NVOC   113
#define EM     85

typedef unsigned int   u32;
typedef unsigned short u16;
typedef __attribute__((ext_vector_type(4)))  u32    u32x4;
typedef __attribute__((ext_vector_type(16))) __bf16 v16bf;
typedef __attribute__((ext_vector_type(8)))  float  v8f;

union Frag { u32x4 u[2]; v16bf v; };

// ---------------- device helpers ----------------
__device__ __forceinline__ u16 f2bf(float f) {
  union { float f; u32 u; } un; un.f = f;
  u32 u = un.u;
  u32 r = u + 0x7FFFu + ((u >> 16) & 1u);   // round-to-nearest-even
  return (u16)(r >> 16);
}
__device__ __forceinline__ float siluf(float x) { return x / (1.f + __expf(-x)); }
__device__ __forceinline__ float softplusf(float x) {
  if (x > 20.f) return x;
  return log1pf(__expf(x));
}

// fragment loaders (ISA 16-bit A 16x32 / B 32x16 wave32 layouts)
__device__ __forceinline__ void loadA(Frag& f, const u16* arow, int k0, int halfw) {
  f.u[0] = *reinterpret_cast<const u32x4*>(arow + k0 + halfw * 8);
  f.u[1] = *reinterpret_cast<const u32x4*>(arow + k0 + 16 + halfw * 8);
}
__device__ __forceinline__ void loadBrow(Frag& f, const u16* brow, int k0, int halfw) {
  const u16* p = brow + k0 + halfw * 8;
  f.u[0] = *reinterpret_cast<const u32x4*>(p);
  f.u[1] = *reinterpret_cast<const u32x4*>(p + 16);
}

// ---------------- WMMA GEMM: C[M,N] = bf16(A[M,K]) @ bf16(W[K,N]) (+bias)(+softplus)
// A: row-major bf16, lda = Kpad (multiple of 32).
// Bt: W transposed, [Npad, Kpad] bf16 (row n = column n of W), Npad mult of 64.
// Each wave: 16 rows x 64 cols (4 accumulators), block = 8 waves = 128 rows.
// K is always a multiple of 64 -> software-pipelined double-buffered fragments.
// launch_bounds(256,1): few waves saturate the matrix pipe; spend VGPRs instead
// of occupancy so the double buffer never spills to scratch.
__global__ void __launch_bounds__(256, 1)
gemm_bf16_kernel(const u16* __restrict__ A, int lda,
                 const u16* __restrict__ Bt, int ldb,
                 float* __restrict__ C, int ldc,
                 const float* __restrict__ bias,
                 int M, int N, int K, int act)
{
  int lane  = threadIdx.x & 31;
  int wave  = threadIdx.x >> 5;
  int l16   = lane & 15;
  int halfw = lane >> 4;                 // 0: lanes 0-15, 1: lanes 16-31
  int row0  = blockIdx.y * 128 + wave * 16;
  int col0  = blockIdx.x * 64;
  if (row0 >= M) return;                  // wave-uniform

  const u16* arow = A + (size_t)(row0 + l16) * lda;
  const u16* br[4];
#pragma unroll
  for (int t = 0; t < 4; ++t)
    br[t] = Bt + (size_t)(col0 + t * 16 + l16) * ldb;

  v8f acc[4];
#pragma unroll
  for (int t = 0; t < 4; ++t)
#pragma unroll
    for (int i = 0; i < 8; ++i) acc[t][i] = 0.f;

  Frag a0, a1, b0[4], b1[4];
  // prologue: stage 0 (k = 0)
  loadA(a0, arow, 0, halfw);
#pragma unroll
  for (int t = 0; t < 4; ++t) loadBrow(b0[t], br[t], 0, halfw);

  for (int k0 = 0; k0 + 64 <= K; k0 += 64) {
    // prefetch one pipeline depth ahead (L2-resident weights/activations)
    if (k0 + 128 < K) {
      __builtin_prefetch(arow + k0 + 128, 0, 3);
      __builtin_prefetch(br[0] + k0 + 128, 0, 3);
    }
    // issue stage-1 loads (k0+32) before consuming stage 0
    loadA(a1, arow, k0 + 32, halfw);
#pragma unroll
    for (int t = 0; t < 4; ++t) loadBrow(b1[t], br[t], k0 + 32, halfw);
    // WMMAs on stage 0
#pragma unroll
    for (int t = 0; t < 4; ++t)
      acc[t] = __builtin_amdgcn_wmma_f32_16x16x32_bf16(
                  false, a0.v, false, b0[t].v, (short)0, acc[t], false, false);
    // issue next stage-0 loads (k0+64) before consuming stage 1
    if (k0 + 64 < K) {
      loadA(a0, arow, k0 + 64, halfw);
#pragma unroll
      for (int t = 0; t < 4; ++t) loadBrow(b0[t], br[t], k0 + 64, halfw);
    }
    // WMMAs on stage 1
#pragma unroll
    for (int t = 0; t < 4; ++t)
      acc[t] = __builtin_amdgcn_wmma_f32_16x16x32_bf16(
                  false, a1.v, false, b1[t].v, (short)0, acc[t], false, false);
  }

  int rbase = row0 + halfw * 8;           // C layout: VGPR i -> row rbase+i
#pragma unroll
  for (int t = 0; t < 4; ++t) {
    int col = col0 + t * 16 + l16;
    if (col >= N) continue;
    float bv = bias ? bias[col] : 0.f;
#pragma unroll
    for (int i = 0; i < 8; ++i) {
      float v = acc[t][i] + bv;
      if (act == 1) v = softplusf(v);
      C[(size_t)(rbase + i) * ldc + col] = v;
    }
  }
}

// ---------------- weight transpose + f32->bf16, zero-padded ----------------
__global__ void prep_wt_kernel(const float* __restrict__ W, int K, int N,
                               u16* __restrict__ Wt, int Kpad, int Npad)
{
  int idx = blockIdx.x * 256 + threadIdx.x;
  int total = Npad * Kpad;
  if (idx >= total) return;
  int n = idx / Kpad, k = idx - n * Kpad;
  float v = (n < N && k < K) ? W[(size_t)k * N + n] : 0.f;
  Wt[idx] = f2bf(v);
}

// ---------------- activation f32 -> bf16 (optional seq flip, K-pad) ----------------
__global__ void cvt_a_kernel(const float* __restrict__ src, int srcLd, int cols,
                             u16* __restrict__ dst, int dstLd, int flip, int total)
{
  int idx = blockIdx.x * 256 + threadIdx.x;
  if (idx >= total) return;
  int r = idx / dstLd, c = idx - r * dstLd;
  int sr = r;
  if (flip) { int b = r >> 10, l = r & 1023; sr = (b << 10) + (1023 - l); }
  float v = (c < cols) ? src[(size_t)sr * srcLd + c] : 0.f;
  dst[idx] = f2bf(v);
}

// ---------------- embedding gather + 3x3 patch conv + sigmoid gate ----------------
// writes flattened (c*9+i*3+j) row of 765, zero-padded to 768, as bf16 A-operand
__global__ void embed_kernel(const int* __restrict__ x, const float* __restrict__ alpha,
                             const float* __restrict__ emb,
                             const float* __restrict__ pw, const float* __restrict__ pb,
                             u16* __restrict__ abf)
{
  int s = blockIdx.x;            // sample 0..4095
  int tid = threadIdx.x;         // 256
  __shared__ float hs[EM * 9];   // [c*9 + i*3 + j]
  __shared__ int   tok[9];
  __shared__ float ga[9];
  if (tid < 9) {
    tok[tid] = x[s * 9 + tid];
    ga[tid] = 1.f / (1.f + __expf(-alpha[tid]));
  }
  __syncthreads();
  for (int t = tid; t < EM * 9; t += 256) {
    int c = t / 9, p = t - c * 9;
    hs[t] = emb[(size_t)tok[p] * EM + c];
  }
  __syncthreads();
  for (int t = tid; t < DM; t += 256) {
    float val = 0.f;
    if (t < EM * 9) {
      int o = t / 9, p = t - o * 9, i = p / 3, j = p - (p / 3) * 3;
      float pe = pb[o];
      for (int c = 0; c < EM; ++c) {
        const float* wr = pw + ((size_t)o * EM + c) * 9;
        const float* hc = hs + c * 9;
#pragma unroll
        for (int di = 0; di < 3; ++di) {
          int ii = i + di - 1;
          if ((unsigned)ii > 2u) continue;
#pragma unroll
          for (int dj = 0; dj < 3; ++dj) {
            int jj = j + dj - 1;
            if ((unsigned)jj > 2u) continue;
            pe += hc[ii * 3 + jj] * wr[di * 3 + dj];
          }
        }
      }
      float g = ga[p];
      val = g * hs[t] + (1.f - g) * pe;
    }
    abf[(size_t)s * DM + t] = f2bf(val);
  }
}

// ---------------- layernorm over 768, optional second (flipped) input ----------------
__global__ void ln_kernel(const float* __restrict__ in1, const float* __restrict__ in2,
                          int flip2, const float* __restrict__ gw,
                          const float* __restrict__ bw, float* __restrict__ out)
{
  int row = blockIdx.x;
  int tid = threadIdx.x;        // 256, 3 cols per thread
  int row2 = row;
  if (flip2) { int b = row >> 10, l = row & 1023; row2 = (b << 10) + (1023 - l); }
  float v[3], s = 0.f, s2 = 0.f;
#pragma unroll
  for (int i = 0; i < 3; ++i) {
    int c = tid + i * 256;
    float xv = in1[(size_t)row * DM + c];
    if (in2) xv += in2[(size_t)row2 * DM + c];
    v[i] = xv; s += xv; s2 += xv * xv;
  }
  __shared__ float sh1[256], sh2[256];
  sh1[tid] = s; sh2[tid] = s2; __syncthreads();
  for (int off = 128; off > 0; off >>= 1) {
    if (tid < off) { sh1[tid] += sh1[tid + off]; sh2[tid] += sh2[tid + off]; }
    __syncthreads();
  }
  float mean = sh1[0] * (1.f / DM);
  float var  = sh2[0] * (1.f / DM) - mean * mean;
  float inv  = rsqrtf(var + 1e-5f);
#pragma unroll
  for (int i = 0; i < 3; ++i) {
    int c = tid + i * 256;
    out[(size_t)row * DM + c] = (v[i] - mean) * inv * gw[c] + bw[c];
  }
}

// ---------------- causal conv1d (width 4) + SiLU; writes f32 and bf16 ----------------
__global__ void conv1d_kernel(const float* __restrict__ xz, const float* __restrict__ cw,
                              const float* __restrict__ cb, float* __restrict__ xc,
                              u16* __restrict__ abf, int total)
{
  int idx = blockIdx.x * 256 + threadIdx.x;
  if (idx >= total) return;
  int d = idx % DI; int r = idx / DI; int b = r >> 10; int l = r & 1023;
  float acc = cb[d];
#pragma unroll
  for (int k = 0; k < DCONV; ++k) {
    int ls = l - (DCONV - 1) + k;
    if (ls >= 0)
      acc += xz[(size_t)((b << 10) + ls) * (2 * DI) + d] * cw[d * DCONV + k];
  }
  acc = siluf(acc);
  xc[idx] = acc;
  abf[idx] = f2bf(acc);
}

// ---------------- selective scan: thread per (batch, d_inner) channel ----------------
// y aliases xc (read x_t, then overwrite same element). D_skip fused.
// B_t / C_t rows fetched as float4 (b128) loads.
__global__ void scan_kernel(const float* __restrict__ dtb, int dtLd,
                            const float* __restrict__ dbc,
                            const float* __restrict__ Alog,
                            const float* __restrict__ Dsk,
                            float* __restrict__ xcy)
{
  int t = blockIdx.x * blockDim.x + threadIdx.x;
  if (t >= BATCH * DI) return;
  int b = t / DI, d = t - b * DI;
  float A[DST];
#pragma unroll
  for (int n = 0; n < DST; ++n) A[n] = -__expf(Alog[d * DST + n]);
  float Dv = Dsk[d];
  float h[DST];
#pragma unroll
  for (int n = 0; n < DST; ++n) h[n] = 0.f;
  for (int l = 0; l < SEQ; ++l) {
    size_t row = (size_t)b * SEQ + l;
    float dt = dtb[row * dtLd + d];
    float xv = xcy[row * DI + d];
    const float4* dB = reinterpret_cast<const float4*>(dbc + row * 80 + DTR);
    float Bv[DST], Cv[DST];
#pragma unroll
    for (int q = 0; q < 4; ++q) {
      float4 vB = dB[q];
      Bv[q * 4 + 0] = vB.x; Bv[q * 4 + 1] = vB.y;
      Bv[q * 4 + 2] = vB.z; Bv[q * 4 + 3] = vB.w;
      float4 vC = dB[q + 4];                 // C block starts 16 floats later
      Cv[q * 4 + 0] = vC.x; Cv[q * 4 + 1] = vC.y;
      Cv[q * 4 + 2] = vC.z; Cv[q * 4 + 3] = vC.w;
    }
    float dx = dt * xv;
    float y = xv * Dv;
#pragma unroll
    for (int n = 0; n < DST; ++n) {
      float dA = __expf(dt * A[n]);
      float hn = h[n] * dA + dx * Bv[n];
      h[n] = hn;
      y += hn * Cv[n];
    }
    xcy[row * DI + d] = y;
  }
}

// ---------------- y * silu(z) -> bf16 A-operand ----------------
__global__ void gate_kernel(const float* __restrict__ y, const float* __restrict__ xz,
                            u16* __restrict__ abf, int total)
{
  int idx = blockIdx.x * 256 + threadIdx.x;
  if (idx >= total) return;
  int d = idx % DI; int r = idx / DI;
  float z = xz[(size_t)r * (2 * DI) + DI + d];
  abf[idx] = f2bf(y[idx] * siluf(z));
}

// ---------------- host orchestration ----------------
struct WsPtrs {
  float* h; float* ta; float* tb;
  u16* abf; u16* wt;
  float* xz; float* xc; float* dbc;
};

static void gemm_go(const u16* A, int lda, const u16* Bt, int ldb,
                    float* C, int ldc, const float* bias,
                    int M, int Npad, int N, int K, int act, hipStream_t st)
{
  dim3 g(Npad / 64, M / 128), b(256);
  gemm_bf16_kernel<<<g, b, 0, st>>>(A, lda, Bt, ldb, C, ldc, bias, M, N, K, act);
}

static void run_mamba(const float* inW, const float* convW, const float* convB,
                      const float* xpW, const float* dtW, const float* dtB,
                      const float* Alog, const float* Dsk, const float* outW,
                      const float* hin, int flip, float* out,
                      const WsPtrs& w, hipStream_t st)
{
  // h (optionally flipped) -> bf16
  { int tot = BL * DM;
    cvt_a_kernel<<<(tot + 255) / 256, 256, 0, st>>>(hin, DM, DM, w.abf, DM, flip, tot); }
  // xz = h @ in_w   [4096,768]x[768,3072]
  { int tot = (2 * DI) * DM;
    prep_wt_kernel<<<(tot + 255) / 256, 256, 0, st>>>(inW, DM, 2 * DI, w.wt, DM, 2 * DI);
    gemm_go(w.abf, DM, w.wt, DM, w.xz, 2 * DI, nullptr, BL, 2 * DI, 2 * DI, DM, 0, st); }
  // causal conv1d + silu
  { int tot = BL * DI;
    conv1d_kernel<<<(tot + 255) / 256, 256, 0, st>>>(w.xz, convW, convB, w.xc, w.abf, tot); }
  // dbc = xc @ xp_w  [4096,1536]x[1536,80]  (Npad 128)
  { int tot = 128 * DI;
    prep_wt_kernel<<<(tot + 255) / 256, 256, 0, st>>>(xpW, DI, DTR + 2 * DST, w.wt, DI, 128);
    gemm_go(w.abf, DI, w.wt, DI, w.dbc, 80, nullptr, BL, 128, 80, DI, 0, st); }
  // dt = softplus(dbc[:, :48] @ dt_w + dt_b), K padded to 64; output over xz cols 0..1535
  { int tot = BL * 64;
    cvt_a_kernel<<<(tot + 255) / 256, 256, 0, st>>>(w.dbc, 80, DTR, w.abf, 64, 0, tot);
    int tw = DI * 64;
    prep_wt_kernel<<<(tw + 255) / 256, 256, 0, st>>>(dtW, DTR, DI, w.wt, 64, DI);
    gemm_go(w.abf, 64, w.wt, 64, w.xz, 2 * DI, dtB, BL, DI, DI, 64, 1, st); }
  // selective scan (y overwrites xc in place)
  scan_kernel<<<(BATCH * DI + 255) / 256, 256, 0, st>>>(w.xz, 2 * DI, w.dbc, Alog, Dsk, w.xc);
  // gate: y * silu(z) -> bf16
  { int tot = BL * DI;
    gate_kernel<<<(tot + 255) / 256, 256, 0, st>>>(w.xc, w.xz, w.abf, tot); }
  // out = yg @ out_w  [4096,1536]x[1536,768]
  { int tot = DM * DI;
    prep_wt_kernel<<<(tot + 255) / 256, 256, 0, st>>>(outW, DI, DM, w.wt, DI, DM);
    gemm_go(w.abf, DI, w.wt, DI, out, DM, nullptr, BL, DM, DM, DI, 0, st); }
}

extern "C" void kernel_launch(void* const* d_in, const int* in_sizes, int n_in,
                              void* d_out, int out_size, void* d_ws, size_t ws_size,
                              hipStream_t stream)
{
  const int*   x      = (const int*)  d_in[0];
  const float* alpha  = (const float*)d_in[1];
  const float* embT   = (const float*)d_in[2];
  const float* patchW = (const float*)d_in[3];
  const float* patchB = (const float*)d_in[4];
  const float* projW  = (const float*)d_in[5];
  const float* projB  = (const float*)d_in[6];
  const float* elnG   = (const float*)d_in[7];
  const float* elnB   = (const float*)d_in[8];
  const float* inW    = (const float*)d_in[9];
  const float* convW  = (const float*)d_in[10];
  const float* convB  = (const float*)d_in[11];
  const float* xpW    = (const float*)d_in[12];
  const float* dtW    = (const float*)d_in[13];
  const float* dtB    = (const float*)d_in[14];
  const float* Alog   = (const float*)d_in[15];
  const float* Dsk    = (const float*)d_in[16];
  const float* outW   = (const float*)d_in[17];
  const float* blnG   = (const float*)d_in[18];
  const float* blnB   = (const float*)d_in[19];
  const float* headW  = (const float*)d_in[20];

  char* ws = (char*)d_ws;
  size_t off = 0;
  auto take = [&](size_t bytes) -> char* {
    char* p = ws + off; off += (bytes + 255) & ~(size_t)255; return p;
  };
  WsPtrs w;
  w.h   = (float*)take((size_t)BL * DM * 4);
  w.ta  = (float*)take((size_t)BL * DM * 4);
  w.tb  = (float*)take((size_t)BL * DM * 4);
  w.abf = (u16*)  take((size_t)BL * DI * 2);
  w.wt  = (u16*)  take((size_t)(2 * DI) * DM * 2);   // largest transposed weight
  w.xz  = (float*)take((size_t)BL * 2 * DI * 4);
  w.xc  = (float*)take((size_t)BL * DI * 4);
  w.dbc = (float*)take((size_t)BL * 80 * 4);
  (void)ws_size; (void)in_sizes; (void)n_in; (void)out_size;

  // ---- embedding: gather + patch conv + gate -> bf16 [4096,768(765 real)]
  embed_kernel<<<BL, 256, 0, stream>>>(x, alpha, embT, patchW, patchB, w.abf);
  // proj: [4096,765]x[765,768] + bias, then LN -> h
  { int tot = DM * DM;
    prep_wt_kernel<<<(tot + 255) / 256, 256, 0, stream>>>(projW, EM * 9, DM, w.wt, DM, DM);
    gemm_go(w.abf, DM, w.wt, DM, w.ta, DM, projB, BL, DM, DM, DM, 0, stream);
    ln_kernel<<<BL, 256, 0, stream>>>(w.ta, nullptr, 0, elnG, elnB, w.h); }

  // layer-parameter offsets
  const size_t oIn = (size_t)DM * 2 * DI, oCw = (size_t)DI * DCONV, oV = DI;
  const size_t oXp = (size_t)DI * (DTR + 2 * DST), oDt = (size_t)DTR * DI;
  const size_t oA = (size_t)DI * DST, oOw = (size_t)DI * DM;

  // bidirectional layer 0
  run_mamba(inW, convW, convB, xpW, dtW, dtB, Alog, Dsk, outW, w.h, 0, w.ta, w, stream);
  run_mamba(inW, convW, convB, xpW, dtW, dtB, Alog, Dsk, outW, w.h, 1, w.tb, w, stream);
  ln_kernel<<<BL, 256, 0, stream>>>(w.ta, w.tb, 1, blnG, blnB, w.h);   // fwd + flip(bwd) -> LN

  // layer 1
  run_mamba(inW + oIn, convW + oCw, convB + oV, xpW + oXp, dtW + oDt, dtB + oV,
            Alog + oA, Dsk + oV, outW + oOw, w.h, 0, w.ta, w, stream);
  // layer 2
  run_mamba(inW + 2 * oIn, convW + 2 * oCw, convB + 2 * oV, xpW + 2 * oXp, dtW + 2 * oDt,
            dtB + 2 * oV, Alog + 2 * oA, Dsk + 2 * oV, outW + 2 * oOw, w.ta, 0, w.tb, w, stream);

  // head: [4096,768]x[768,113] -> d_out (Npad 128)
  { int tot = BL * DM;
    cvt_a_kernel<<<(tot + 255) / 256, 256, 0, stream>>>(w.tb, DM, DM, w.abf, DM, 0, tot);
    int tw = 128 * DM;
    prep_wt_kernel<<<(tw + 255) / 256, 256, 0, stream>>>(headW, DM, NVOC, w.wt, DM, 128);
    gemm_go(w.abf, DM, w.wt, DM, (float*)d_out, NVOC, nullptr, BL, 128, NVOC, DM, 0, stream); }
}
